// Adapter_Router_plus_65180423684549
// MI455X (gfx1250) — matrix operand order repaired and checked by reading, not compiled, not run on hardware
//
#include <hip/hip_runtime.h>
#include <cstdint>
#include <cstddef>

// Problem constants (match reference)
#define B_  8
#define N_  2048
#define C_  4096
#define E_  2
#define H_  64
#define G_  2
#define HG_ (H_ / G_)   // 32
#define CG_ (C_ / G_)   // 2048
#define SCALE_ 1.0f
#define T_     10.0f

#define NT_ 32          // rows (tokens) per block (two 16-row WMMA tiles)
#define KC_ 128         // K chunk staged in LDS per iteration

typedef __attribute__((ext_vector_type(16))) __bf16 v16bf;
typedef __attribute__((ext_vector_type(2)))  __bf16 v2bf;
typedef __attribute__((ext_vector_type(8)))  float  v8f;

union FragBF {
    v16bf bf;
    uint4 u[2];
};

__device__ __forceinline__ unsigned pack_bf16(float a, float b) {
#if __has_builtin(__builtin_amdgcn_cvt_pk_bf16_f32)
    union { v2bf v; unsigned u; } p;
    p.v = __builtin_amdgcn_cvt_pk_bf16_f32(a, b);
    return p.u;
#else
    union { __bf16 h[2]; unsigned u; } p;
    p.h[0] = (__bf16)a;
    p.h[1] = (__bf16)b;
    return p.u;
#endif
}

// ---------------------------------------------------------------------------
// Kernel 1: router  w[b,e] = softmax((x[b,0,:] @ We[e,:] + be[e]) / T)
// ---------------------------------------------------------------------------
__global__ __launch_bounds__(256) void router_kernel(
    const float* __restrict__ x, const float* __restrict__ We,
    const float* __restrict__ be, float* __restrict__ w_out) {
    const int b   = blockIdx.x;
    const int tid = threadIdx.x;
    const float* xr = x + (size_t)b * N_ * C_;   // token 0 of sample b
    __shared__ float red0[256];
    __shared__ float red1[256];
    float s0 = 0.f, s1 = 0.f;
    for (int c = tid; c < C_; c += 256) {
        const float xv = xr[c];
        s0 += xv * We[c];
        s1 += xv * We[C_ + c];
    }
    red0[tid] = s0; red1[tid] = s1;
    __syncthreads();
    for (int off = 128; off > 0; off >>= 1) {
        if (tid < off) { red0[tid] += red0[tid + off]; red1[tid] += red1[tid + off]; }
        __syncthreads();
    }
    if (tid == 0) {
        const float l0 = (red0[0] + be[0]) / T_;
        const float l1 = (red1[0] + be[1]) / T_;
        const float m  = fmaxf(l0, l1);
        const float e0 = __expf(l0 - m);
        const float e1 = __expf(l1 - m);
        const float inv = 1.f / (e0 + e1);
        w_out[b * 2 + 0] = e0 * inv;
        w_out[b * 2 + 1] = e1 * inv;
    }
}

// ---------------------------------------------------------------------------
// Kernel 2: f32 -> bf16 weight conversion (Wa, Wb are tiny, L2-resident)
// ---------------------------------------------------------------------------
__global__ __launch_bounds__(256) void cvt_bf16_kernel(
    const float* __restrict__ src, __bf16* __restrict__ dst, int n) {
    const int i = blockIdx.x * 256 + threadIdx.x;
    if (i < n) dst[i] = (__bf16)src[i];
}

// ---------------------------------------------------------------------------
// Kernel 3: fused down-proj (silu) + grouped up-proj + router-weighted
//           combine + bias + residual.   One block = 32 tokens of sample b.
// ---------------------------------------------------------------------------
__global__ __launch_bounds__(256) void fused_adapter_kernel(
    const float*  __restrict__ x,
    const float*  __restrict__ ba,
    const float*  __restrict__ bb,
    const float*  __restrict__ wrt,     // router weights [B,E]
    const __bf16* __restrict__ WaB,     // [E,H,C] bf16
    const __bf16* __restrict__ WbB,     // [E,G,C/G,H/G] bf16
    float*        __restrict__ out) {

    const int b   = blockIdx.y;
    const int n0  = blockIdx.x * NT_;
    const int tid = threadIdx.x;
    const int wv  = tid >> 5;        // wave 0..7
    const int ln  = tid & 31;
    const int hf  = ln >> 4;         // lane half (0/1)
    const int cl  = ln & 15;         // lane column

    __shared__ __attribute__((aligned(16))) __bf16 Xs[NT_][KC_];      // 8 KB
    __shared__ __attribute__((aligned(16))) __bf16 Ys[E_][NT_][H_];   // 8 KB

    const float w0s = wrt[b * 2 + 0] * SCALE_;
    const float w1s = wrt[b * 2 + 1] * SCALE_;

    // ---------------- Phase 1: Y[e] = silu(X @ Wa[e]^T + ba[e]) -----------
    // wave -> (expert, h-tile): 8 waves cover E(2) x Htiles(4).
    // Each wave computes TWO 16-row tiles sharing the same B (weight) frags.
    const int e1 = wv >> 2;
    const int ht = wv & 3;

    v8f acc0 = {0.f, 0.f, 0.f, 0.f, 0.f, 0.f, 0.f, 0.f};
    v8f acc1 = {0.f, 0.f, 0.f, 0.f, 0.f, 0.f, 0.f, 0.f};

    const float* xrow = x + ((size_t)b * N_ + n0) * C_;
    const int srow  = tid >> 3;         // staging row 0..31
    const int scolf = (tid & 7) * 16;   // staging col base (16 floats/thread)

    for (int kc = 0; kc < C_; kc += KC_) {
        // cooperative stage: X[32][128] f32 -> bf16 in LDS
        const float* p = xrow + (size_t)srow * C_ + kc + scolf;
        const float4 a0 = *(const float4*)p;
        const float4 a1 = *(const float4*)(p + 4);
        const float4 a2 = *(const float4*)(p + 8);
        const float4 a3 = *(const float4*)(p + 12);
        uint4 pk0, pk1;
        pk0.x = pack_bf16(a0.x, a0.y);
        pk0.y = pack_bf16(a0.z, a0.w);
        pk0.z = pack_bf16(a1.x, a1.y);
        pk0.w = pack_bf16(a1.z, a1.w);
        pk1.x = pack_bf16(a2.x, a2.y);
        pk1.y = pack_bf16(a2.z, a2.w);
        pk1.z = pack_bf16(a3.x, a3.y);
        pk1.w = pack_bf16(a3.z, a3.w);
        *(uint4*)&Xs[srow][scolf]     = pk0;
        *(uint4*)&Xs[srow][scolf + 8] = pk1;
        if (kc + KC_ < C_) {
            __builtin_prefetch(p + KC_, 0, 1);  // global_prefetch next chunk
        }
        __syncthreads();

        const int hch = ht * 16 + cl;   // this lane's output channel
#pragma unroll
        for (int ks = 0; ks < KC_; ks += 32) {
            // B fragment (32x16 bf16): lane holds 16 contiguous K values
            FragBF fb;
            const __bf16* wp = WaB + ((size_t)(e1 * H_ + hch)) * C_ + kc + ks + hf * 16;
            fb.u[0] = *(const uint4*)wp;
            fb.u[1] = *(const uint4*)(wp + 8);
            // A fragments (16x32 bf16): lane<16 holds K {0..15}
            const int k0 = ks + hf * 16;
            FragBF fa0, fa1;
            fa0.u[0] = *(const uint4*)&Xs[cl][k0];
            fa0.u[1] = *(const uint4*)&Xs[cl][k0 + 8];
            fa1.u[0] = *(const uint4*)&Xs[16 + cl][k0];
            fa1.u[1] = *(const uint4*)&Xs[16 + cl][k0 + 8];
            acc0 = __builtin_amdgcn_wmma_f32_16x16x32_bf16(
                false, fa0.bf, false, fb.bf, (short)0, acc0, false, false);
            acc1 = __builtin_amdgcn_wmma_f32_16x16x32_bf16(
                false, fa1.bf, false, fb.bf, (short)0, acc1, false, false);
        }
        __syncthreads();
    }

    // silu + fold router weight, store Y tiles to LDS as bf16
    {
        const int hch = ht * 16 + cl;
        const float bav = ba[e1 * H_ + hch];
        const float wsc = (e1 == 0) ? w0s : w1s;
#pragma unroll
        for (int r = 0; r < 8; ++r) {
            const float v0 = acc0[r] + bav;
            const float v1 = acc1[r] + bav;
            const float s0 = v0 / (1.f + __expf(-v0));   // silu
            const float s1 = v1 / (1.f + __expf(-v1));
            Ys[e1][r + hf * 8][hch]      = (__bf16)(s0 * wsc);
            Ys[e1][16 + r + hf * 8][hch] = (__bf16)(s1 * wsc);
        }
    }
    __syncthreads();

    // ---------------- Phase 2: out = (Yw @ Wb) + bias + x -----------------
    // 256 column tiles of 16; wave wv owns tiles [wv*32, wv*32+32) -> one group.
    const int g = wv >> 2;   // tiles 0..127 are group 0, 128..255 group 1

    // A fragments per (expert, row-tile); K = 32 = H/G, constant for this wave
    FragBF fa00, fa01, fa10, fa11;   // fa[e][rowtile]
    {
        const int kb = g * HG_ + hf * 16;
        fa00.u[0] = *(const uint4*)&Ys[0][cl][kb];
        fa00.u[1] = *(const uint4*)&Ys[0][cl][kb + 8];
        fa01.u[0] = *(const uint4*)&Ys[0][16 + cl][kb];
        fa01.u[1] = *(const uint4*)&Ys[0][16 + cl][kb + 8];
        fa10.u[0] = *(const uint4*)&Ys[1][cl][kb];
        fa10.u[1] = *(const uint4*)&Ys[1][cl][kb + 8];
        fa11.u[0] = *(const uint4*)&Ys[1][16 + cl][kb];
        fa11.u[1] = *(const uint4*)&Ys[1][16 + cl][kb + 8];
    }

    const float* xres = x   + ((size_t)b * N_ + n0) * C_;
    float*       orow = out + ((size_t)b * N_ + n0) * C_;

    for (int i = 0; i < 32; ++i) {
        const int cb = wv * 512 + i * 16;   // column base within C
        const int cp = cb & (CG_ - 1);      // column within group
        const int cc = cb + cl;             // this lane's column

        // B fragments: Wb[e][g][c'][h'], K(h') contiguous -> 32B per lane.
        // Each B fragment feeds BOTH row tiles (2 wmmas per fetch).
        FragBF fb0, fb1;
        const __bf16* wb0 = WbB + (((size_t)(0 * G_ + g) * CG_ + cp + cl)) * HG_ + hf * 16;
        const __bf16* wb1 = WbB + (((size_t)(1 * G_ + g) * CG_ + cp + cl)) * HG_ + hf * 16;
        fb0.u[0] = *(const uint4*)wb0;
        fb0.u[1] = *(const uint4*)(wb0 + 8);
        fb1.u[0] = *(const uint4*)wb1;
        fb1.u[1] = *(const uint4*)(wb1 + 8);

        v8f z0 = {0.f, 0.f, 0.f, 0.f, 0.f, 0.f, 0.f, 0.f};
        v8f z1 = {0.f, 0.f, 0.f, 0.f, 0.f, 0.f, 0.f, 0.f};
        z0 = __builtin_amdgcn_wmma_f32_16x16x32_bf16(
            false, fa00.bf, false, fb0.bf, (short)0, z0, false, false);
        z0 = __builtin_amdgcn_wmma_f32_16x16x32_bf16(
            false, fa10.bf, false, fb1.bf, (short)0, z0, false, false);
        z1 = __builtin_amdgcn_wmma_f32_16x16x32_bf16(
            false, fa01.bf, false, fb0.bf, (short)0, z1, false, false);
        z1 = __builtin_amdgcn_wmma_f32_16x16x32_bf16(
            false, fa11.bf, false, fb1.bf, (short)0, z1, false, false);

        const float biasc = w0s * bb[cc] + w1s * bb[C_ + cc];
#pragma unroll
        for (int r = 0; r < 8; ++r) {
            const int m0 = r + hf * 8;
            const int m1 = 16 + r + hf * 8;
            orow[(size_t)m0 * C_ + cc] = z0[r] + biasc + xres[(size_t)m0 * C_ + cc];
            orow[(size_t)m1 * C_ + cc] = z1[r] + biasc + xres[(size_t)m1 * C_ + cc];
        }
    }
}

// ---------------------------------------------------------------------------
// Launch: inputs (x, Wa, ba, Wb, bb, We, be) all float32
// ---------------------------------------------------------------------------
extern "C" void kernel_launch(void* const* d_in, const int* in_sizes, int n_in,
                              void* d_out, int out_size, void* d_ws, size_t ws_size,
                              hipStream_t stream) {
    (void)in_sizes; (void)n_in; (void)out_size; (void)ws_size;
    const float* x  = (const float*)d_in[0];
    const float* Wa = (const float*)d_in[1];
    const float* ba = (const float*)d_in[2];
    const float* Wb = (const float*)d_in[3];
    const float* bb = (const float*)d_in[4];
    const float* We = (const float*)d_in[5];
    const float* be = (const float*)d_in[6];
    float* out = (float*)d_out;

    // workspace layout: [router w: 256B] [Wa bf16: 1MB] [Wb bf16: 512KB]
    float*  d_w   = (float*)d_ws;
    __bf16* d_WaB = (__bf16*)((char*)d_ws + 256);
    __bf16* d_WbB = (__bf16*)((char*)d_ws + 256 + sizeof(__bf16) * (size_t)E_ * H_ * C_);

    router_kernel<<<B_, 256, 0, stream>>>(x, We, be, d_w);

    const int nWa = E_ * H_ * C_;            // 524288
    const int nWb = E_ * G_ * CG_ * HG_;     // 262144
    cvt_bf16_kernel<<<(nWa + 255) / 256, 256, 0, stream>>>(Wa, d_WaB, nWa);
    cvt_bf16_kernel<<<(nWb + 255) / 256, 256, 0, stream>>>(Wb, d_WbB, nWb);

    fused_adapter_kernel<<<dim3(N_ / NT_, B_), 256, 0, stream>>>(
        x, ba, bb, d_w, d_WaB, d_WbB, out);
}
